// TransformerLayer_13237089206673
// MI455X (gfx1250) — matrix operand order repaired and checked by reading, not compiled, not run on hardware
//
#include <hip/hip_runtime.h>
#include <cstdint>
#include <cstddef>

// ---------------------------------------------------------------------------
// CDNA5 (gfx1250) wave32 WMMA transformer layer.
// GEMMs use v_wmma_f32_16x16x32_bf16; A/K tiles staged with
// global_load_async_to_lds_b128 (ASYNCcnt); softmax / LN in fp32.
// ---------------------------------------------------------------------------

typedef __attribute__((ext_vector_type(16))) __bf16 v16bf;
typedef __attribute__((ext_vector_type(8)))  float  v8f;

union Frag16 { v16bf v; unsigned int u[8]; };

__device__ __forceinline__ unsigned short f2bf(float f) {
  unsigned int u = __builtin_bit_cast(unsigned int, f);
  u += 0x7fffu + ((u >> 16) & 1u);     // round-to-nearest-even
  return (unsigned short)(u >> 16);
}

__device__ __forceinline__ v8f wmma_bf16(v16bf a, v16bf b, v8f c) {
  // (neg_a, A, neg_b, B, c_mod, C, reuse_a, reuse_b)
  return __builtin_amdgcn_wmma_f32_16x16x32_bf16(false, a, false, b, (short)0, c,
                                                 false, false);
}

__device__ __forceinline__ v8f vzero8() {
  v8f z = {0.f, 0.f, 0.f, 0.f, 0.f, 0.f, 0.f, 0.f};
  return z;
}

// Wave-relative LDS byte offset: low 32 bits of the flat address of a
// __shared__ object are the LDS offset (ISA 10.2 aperture mapping).
__device__ __forceinline__ unsigned lds_off(const void* p) {
  return (unsigned)(unsigned long long)p;
}

// fp32 -> bf16 conversion kernel
__global__ void cvt_bf16_kernel(const float* __restrict__ src,
                                unsigned short* __restrict__ dst, int n) {
  int i = blockIdx.x * 256 + threadIdx.x;
  if (i < n) dst[i] = f2bf(src[i]);
}

// ---------------------------------------------------------------------------
// GEMM: out[M,N] = A[M,K](bf16) @ W[K,N](bf16) + bias, optional ReLU.
// Block tile 128x64, 8 waves (4x2 grid of 32x32 wave tiles), K step 32.
// A tile copied global->LDS with async b128 (no transpose needed).
// B tile staged transposed (Bl[n][k]) so fragment K-pairs are dword loads.
// ---------------------------------------------------------------------------
#define LDA 40   // ushorts; 80B rows -> 16B-aligned async stores, no bank dup
#define LDB 34

__global__ __launch_bounds__(256) void gemm_bias_kernel(
    const unsigned short* __restrict__ A, const unsigned short* __restrict__ W,
    const float* __restrict__ bias, float* __restrict__ outF,
    unsigned short* __restrict__ outB, int M, int N, int K, int doRelu) {
  __shared__ alignas(16) unsigned short Al[128 * LDA];
  __shared__ alignas(16) unsigned short Bl[64 * LDB];

  const int t    = threadIdx.x;
  const int lane = t & 31;
  const int wv   = t >> 5;
  const int wm   = (wv & 3) * 32;   // wave row base within tile
  const int wn   = (wv >> 2) * 32;  // wave col base within tile
  const int ln   = lane & 15;
  const int lh   = lane >> 4;       // 0/1 half of wave
  const int rowBase = blockIdx.y * 128;
  const int colBase = blockIdx.x * 64;

  v8f c[2][2];
  c[0][0] = vzero8(); c[0][1] = vzero8();
  c[1][0] = vzero8(); c[1][1] = vzero8();

  const int arow = t >> 1;          // 0..127
  const int acol = (t & 1) * 16;    // 0/16 ushorts
  const int bk   = t >> 3;          // 0..31
  const int bseg = (t & 7) * 8;     // 0..56

  const unsigned short* aPtr = A + (size_t)(rowBase + arow) * K + acol;
  const unsigned short* bPtr = W + (size_t)bk * N + colBase + bseg;
  const unsigned aLds = lds_off(Al + arow * LDA + acol);

  for (int kc = 0; kc < K; kc += 32) {
    __syncthreads();
    // A tile 128x32: async copy global -> LDS (2x b128 per thread)
    {
      const unsigned short* gp = aPtr + kc;
      asm volatile(
          "global_load_async_to_lds_b128 %0, %1, off\n\t"
          "global_load_async_to_lds_b128 %0, %1, off offset:16"
          :: "v"(aLds), "v"(gp) : "memory");
    }
    // B tile 32x64 -> transposed LDS (Bl[n][k])
    {
      uint4 bvx = *(const uint4*)(bPtr + (size_t)kc * N);
      const unsigned short* bs = (const unsigned short*)&bvx;
#pragma unroll
      for (int i = 0; i < 8; ++i) Bl[(bseg + i) * LDB + bk] = bs[i];
    }
    if (kc + 32 < K) {
      __builtin_prefetch(aPtr + kc + 32, 0, 1);          // global_prefetch_b8
      __builtin_prefetch(bPtr + (size_t)(kc + 32) * N, 0, 1);
    }
    asm volatile("s_wait_asynccnt 0x0" ::: "memory");
    __syncthreads();

    Frag16 af[2], bf[2];
#pragma unroll
    for (int f = 0; f < 2; ++f) {
      const unsigned short* base = Al + (wm + f * 16 + ln) * LDA;
#pragma unroll
      for (int j = 0; j < 8; ++j) {
        int kk = 2 * j + lh * 16;
        af[f].u[j] = *(const unsigned int*)(base + kk);
      }
    }
#pragma unroll
    for (int f = 0; f < 2; ++f) {
      const unsigned short* base = Bl + (wn + f * 16 + ln) * LDB + lh * 16;
#pragma unroll
      for (int j = 0; j < 8; ++j)
        bf[f].u[j] = *(const unsigned int*)(base + 2 * j);
    }
#pragma unroll
    for (int i = 0; i < 2; ++i)
#pragma unroll
      for (int j2 = 0; j2 < 2; ++j2)
        c[i][j2] = wmma_bf16(af[i].v, bf[j2].v, c[i][j2]);
  }

  // epilogue: bias (+relu), store f32 and/or bf16
  float bcol[2];
  bcol[0] = bias[colBase + wn + ln];
  bcol[1] = bias[colBase + wn + 16 + ln];
#pragma unroll
  for (int i = 0; i < 2; ++i) {
#pragma unroll
    for (int j2 = 0; j2 < 2; ++j2) {
      int col = colBase + wn + j2 * 16 + ln;
#pragma unroll
      for (int r = 0; r < 8; ++r) {
        float vv = c[i][j2][r] + bcol[j2];
        if (doRelu) vv = fmaxf(vv, 0.f);
        int row = rowBase + wm + i * 16 + r + lh * 8;
        size_t idx = (size_t)row * N + col;
        if (outF) outF[idx] = vv;
        if (outB) outB[idx] = f2bf(vv);
      }
    }
  }
}

// ---------------------------------------------------------------------------
// Flash attention: grid = (S/64, B*H), 4 waves/block, each wave 16 queries.
// Streams causal 64-key blocks; online softmax in fp32; QK^T and P@V via WMMA.
// K tile staged with async b128 copies; V tile transposed manually.
// D=512, H=8, DK=64, scale = 1/sqrt(64) = 0.125.
// ---------------------------------------------------------------------------
#define KSD 72   // ushorts; 144B rows -> 16B-aligned async stores
#define VSD 66

__global__ __launch_bounds__(128) void attn_kernel(
    const unsigned short* __restrict__ Q, const unsigned short* __restrict__ Kb,
    const unsigned short* __restrict__ Vb, unsigned short* __restrict__ Ctx,
    int S) {
  __shared__ alignas(16) unsigned short Kl[64 * KSD];      // K[key][dk]
  __shared__ alignas(16) unsigned short Vt[64 * VSD];      // V^T[dk][key]
  __shared__ alignas(16) unsigned short Pl[4][16 * 64];    // per-wave probs

  const int t    = threadIdx.x;
  const int lane = t & 31;
  const int w    = t >> 5;
  const int ln   = lane & 15;
  const int lh   = lane >> 4;

  const int bh   = blockIdx.y;
  const int b    = bh >> 3;
  const int h    = bh & 7;
  const int qblk = blockIdx.x;
  const int q0   = qblk * 64 + w * 16;
  const size_t rowOff = (size_t)b * S;
  const int hoff = h * 64;

  // preload Q fragments for both 32-wide k-steps of DK=64
  Frag16 aq[2];
  {
    const unsigned short* qrow = Q + (rowOff + q0 + ln) * 512 + hoff;
#pragma unroll
    for (int ks = 0; ks < 2; ++ks)
#pragma unroll
      for (int j = 0; j < 8; ++j) {
        int dk = ks * 32 + 2 * j + lh * 16;
        aq[ks].u[j] = *(const unsigned int*)(qrow + dk);
      }
  }

  v8f o[4];
  o[0] = vzero8(); o[1] = vzero8(); o[2] = vzero8(); o[3] = vzero8();
  float mrun[8], lrun[8];
#pragma unroll
  for (int r = 0; r < 8; ++r) { mrun[r] = -3.0e38f; lrun[r] = 0.f; }

  const int krow  = t >> 1;          // 0..63
  const int khalf = (t & 1) * 32;    // 0/32 ushorts
  const unsigned kLds = lds_off(Kl + krow * KSD + khalf);

  for (int kbI = 0; kbI <= qblk; ++kbI) {
    __syncthreads();
    // stage K (async, direct) and V (manual, transposed) 64x64 bf16 tiles
    {
      const unsigned short* kp =
          Kb + (rowOff + kbI * 64 + krow) * 512 + hoff + khalf;
      asm volatile(
          "global_load_async_to_lds_b128 %0, %1, off\n\t"
          "global_load_async_to_lds_b128 %0, %1, off offset:16\n\t"
          "global_load_async_to_lds_b128 %0, %1, off offset:32\n\t"
          "global_load_async_to_lds_b128 %0, %1, off offset:48"
          :: "v"(kLds), "v"(kp) : "memory");

      const unsigned short* vp =
          Vb + (rowOff + kbI * 64 + krow) * 512 + hoff + khalf;
      const uint4* vp4 = (const uint4*)vp;
#pragma unroll
      for (int q = 0; q < 4; ++q) {
        uint4 u = vp4[q];
        const unsigned short* us = (const unsigned short*)&u;
#pragma unroll
        for (int i = 0; i < 8; ++i)
          Vt[(khalf + q * 8 + i) * VSD + krow] = us[i];
      }
    }
    asm volatile("s_wait_asynccnt 0x0" ::: "memory");
    __syncthreads();

    // scores S = Q K^T  (4 column tiles of 16 keys)
    v8f s[4];
#pragma unroll
    for (int cc = 0; cc < 4; ++cc) {
      s[cc] = vzero8();
#pragma unroll
      for (int ks = 0; ks < 2; ++ks) {
        Frag16 bf;
        const unsigned short* base = Kl + (cc * 16 + ln) * KSD + ks * 32 + lh * 16;
#pragma unroll
        for (int j = 0; j < 8; ++j)
          bf.u[j] = *(const unsigned int*)(base + 2 * j);
        s[cc] = wmma_bf16(aq[ks].v, bf.v, s[cc]);
      }
    }

    // scale + causal mask
    const int colg0 = kbI * 64;
    const bool needMask = (colg0 + 63) > q0;
    float cmax[8];
#pragma unroll
    for (int r = 0; r < 8; ++r) cmax[r] = -3.0e38f;
#pragma unroll
    for (int cc = 0; cc < 4; ++cc) {
#pragma unroll
      for (int r = 0; r < 8; ++r) {
        float vv = s[cc][r] * 0.125f;
        if (needMask) {
          int keyg = colg0 + cc * 16 + ln;
          int rowg = q0 + r + lh * 8;
          if (keyg > rowg) vv = -1.0e9f;
        }
        s[cc][r] = vv;
        cmax[r] = fmaxf(cmax[r], vv);
      }
    }
    // row max across the 16 lanes owning this row (halves stay separate)
#pragma unroll
    for (int r = 0; r < 8; ++r) {
#pragma unroll
      for (int mk = 1; mk <= 8; mk <<= 1)
        cmax[r] = fmaxf(cmax[r], __shfl_xor(cmax[r], mk, 32));
    }
    float ladd[8];
#pragma unroll
    for (int r = 0; r < 8; ++r) {
      float mn = fmaxf(mrun[r], cmax[r]);
      float al = __expf(mrun[r] - mn);
      mrun[r] = mn;
      lrun[r] *= al;
      ladd[r] = 0.f;
#pragma unroll
      for (int cc = 0; cc < 4; ++cc) o[cc][r] *= al;
    }
#pragma unroll
    for (int cc = 0; cc < 4; ++cc) {
#pragma unroll
      for (int r = 0; r < 8; ++r) {
        float p = __expf(s[cc][r] - mrun[r]);
        s[cc][r] = p;
        ladd[r] += p;
      }
    }
#pragma unroll
    for (int r = 0; r < 8; ++r) {
#pragma unroll
      for (int mk = 1; mk <= 8; mk <<= 1)
        ladd[r] += __shfl_xor(ladd[r], mk, 32);
      lrun[r] += ladd[r];
    }

    // stage probabilities (bf16) for the P@V WMMA A-fragments
#pragma unroll
    for (int cc = 0; cc < 4; ++cc)
#pragma unroll
      for (int r = 0; r < 8; ++r)
        Pl[w][(r + lh * 8) * 64 + cc * 16 + ln] = f2bf(s[cc][r]);
    __syncthreads();

    // O += P @ V  (4 dk column tiles, 2 key-steps of 32)
#pragma unroll
    for (int cd = 0; cd < 4; ++cd) {
#pragma unroll
      for (int ks = 0; ks < 2; ++ks) {
        Frag16 ap, bvf;
#pragma unroll
        for (int j = 0; j < 8; ++j) {
          int key = ks * 32 + 2 * j + lh * 16;
          ap.u[j] = *(const unsigned int*)(&Pl[w][ln * 64 + key]);
        }
        const unsigned short* vbase = Vt + (cd * 16 + ln) * VSD + ks * 32 + lh * 16;
#pragma unroll
        for (int j = 0; j < 8; ++j)
          bvf.u[j] = *(const unsigned int*)(vbase + 2 * j);
        o[cd] = wmma_bf16(ap.v, bvf.v, o[cd]);
      }
    }
  }

  // normalize and write ctx (bf16, head-interleaved back to [B,S,D])
#pragma unroll
  for (int r = 0; r < 8; ++r) {
    float inv = 1.f / lrun[r];
#pragma unroll
    for (int cd = 0; cd < 4; ++cd) {
      int row = q0 + r + lh * 8;
      int col = hoff + cd * 16 + ln;
      Ctx[(rowOff + row) * 512 + col] = f2bf(o[cd][r] * inv);
    }
  }
}

// ---------------------------------------------------------------------------
// Residual add + LayerNorm over D=512. One row per block, 128 threads.
// ---------------------------------------------------------------------------
__global__ __launch_bounds__(128) void add_ln_kernel(
    const float* __restrict__ X, const float* __restrict__ A,
    const float* __restrict__ G, const float* __restrict__ Bb,
    float* __restrict__ outF, unsigned short* __restrict__ outB) {
  __shared__ float ws[4], wsq[4];
  const int row = blockIdx.x;
  const int t   = threadIdx.x;
  const float* px = X + (size_t)row * 512;
  const float* pa = A + (size_t)row * 512;
  float v[4];
  float s = 0.f, sq = 0.f;
#pragma unroll
  for (int i = 0; i < 4; ++i) {
    float vv = px[t + i * 128] + pa[t + i * 128];
    v[i] = vv; s += vv; sq += vv * vv;
  }
#pragma unroll
  for (int mk = 1; mk <= 16; mk <<= 1) {
    s  += __shfl_xor(s, mk, 32);
    sq += __shfl_xor(sq, mk, 32);
  }
  if ((t & 31) == 0) { ws[t >> 5] = s; wsq[t >> 5] = sq; }
  __syncthreads();
  float St  = ws[0] + ws[1] + ws[2] + ws[3];
  float SQt = wsq[0] + wsq[1] + wsq[2] + wsq[3];
  float mean = St * (1.f / 512.f);
  float var  = SQt * (1.f / 512.f) - mean * mean;
  float rstd = rsqrtf(var + 1e-5f);
#pragma unroll
  for (int i = 0; i < 4; ++i) {
    int idx = t + i * 128;
    float y = (v[i] - mean) * rstd * G[idx] + Bb[idx];
    size_t o = (size_t)row * 512 + idx;
    if (outF) outF[o] = y;
    if (outB) outB[o] = f2bf(y);
  }
}

// ---------------------------------------------------------------------------
// Host orchestration
// ---------------------------------------------------------------------------
extern "C" void kernel_launch(void* const* d_in, const int* in_sizes, int n_in,
                              void* d_out, int out_size, void* d_ws,
                              size_t ws_size, hipStream_t stream) {
  (void)in_sizes; (void)n_in; (void)out_size; (void)ws_size;
  constexpr int Bn = 4, S = 2048, D = 512, FF = 2048;
  constexpr int M = Bn * S;

  const float* x   = (const float*)d_in[0];
  const float* wq  = (const float*)d_in[2];
  const float* bq  = (const float*)d_in[3];
  const float* wk  = (const float*)d_in[4];
  const float* bk  = (const float*)d_in[5];
  const float* wv  = (const float*)d_in[6];
  const float* bv  = (const float*)d_in[7];
  const float* wo  = (const float*)d_in[8];
  const float* bo  = (const float*)d_in[9];
  const float* w1  = (const float*)d_in[10];
  const float* b1  = (const float*)d_in[11];
  const float* w2  = (const float*)d_in[12];
  const float* b2  = (const float*)d_in[13];
  const float* g1  = (const float*)d_in[14];
  const float* be1 = (const float*)d_in[15];
  const float* g2  = (const float*)d_in[16];
  const float* be2 = (const float*)d_in[17];

  char* ws = (char*)d_ws;
  size_t off = 0;
  auto alloc = [&](size_t bytes) -> void* {
    void* p = ws + off;
    off += (bytes + 255) & ~(size_t)255;
    return p;
  };

  unsigned short* xb   = (unsigned short*)alloc((size_t)M * D * 2);
  unsigned short* wqb  = (unsigned short*)alloc((size_t)D * D * 2);
  unsigned short* wkb  = (unsigned short*)alloc((size_t)D * D * 2);
  unsigned short* wvb  = (unsigned short*)alloc((size_t)D * D * 2);
  unsigned short* wob  = (unsigned short*)alloc((size_t)D * D * 2);
  unsigned short* w1b  = (unsigned short*)alloc((size_t)D * FF * 2);
  unsigned short* w2b  = (unsigned short*)alloc((size_t)FF * D * 2);
  unsigned short* qb   = (unsigned short*)alloc((size_t)M * D * 2);
  unsigned short* kbuf = (unsigned short*)alloc((size_t)M * D * 2);
  unsigned short* vbuf = (unsigned short*)alloc((size_t)M * D * 2);
  unsigned short* ctxb = (unsigned short*)alloc((size_t)M * D * 2);
  float*          attf = (float*)alloc((size_t)M * D * 4);
  float*          h_f  = (float*)alloc((size_t)M * D * 4);
  unsigned short* h_b  = (unsigned short*)alloc((size_t)M * D * 2);
  // aliases (lifetimes disjoint): relu acts reuse q/k/v/ctx (exactly 32 MiB);
  // ffn output reuses attn_out.
  unsigned short* r1b  = qb;
  float*          ffnf = attf;

  auto cvt = [&](const float* s, unsigned short* d, int n) {
    cvt_bf16_kernel<<<(n + 255) / 256, 256, 0, stream>>>(s, d, n);
  };
  cvt(x, xb, M * D);
  cvt(wq, wqb, D * D); cvt(wk, wkb, D * D);
  cvt(wv, wvb, D * D); cvt(wo, wob, D * D);
  cvt(w1, w1b, D * FF); cvt(w2, w2b, FF * D);

  dim3 blk(256);
  // Q, K, V projections (bf16 outputs only)
  gemm_bias_kernel<<<dim3(D / 64, M / 128), blk, 0, stream>>>(
      xb, wqb, bq, nullptr, qb, M, D, D, 0);
  gemm_bias_kernel<<<dim3(D / 64, M / 128), blk, 0, stream>>>(
      xb, wkb, bk, nullptr, kbuf, M, D, D, 0);
  gemm_bias_kernel<<<dim3(D / 64, M / 128), blk, 0, stream>>>(
      xb, wvb, bv, nullptr, vbuf, M, D, D, 0);

  // flash attention (causal)
  attn_kernel<<<dim3(S / 64, Bn * 8), dim3(128), 0, stream>>>(
      qb, kbuf, vbuf, ctxb, S);

  // output projection
  gemm_bias_kernel<<<dim3(D / 64, M / 128), blk, 0, stream>>>(
      ctxb, wob, bo, attf, nullptr, M, D, D, 0);

  // residual + LN1 (f32 + bf16 copies)
  add_ln_kernel<<<M, 128, 0, stream>>>(x, attf, g1, be1, h_f, h_b);

  // FFN
  gemm_bias_kernel<<<dim3(FF / 64, M / 128), blk, 0, stream>>>(
      h_b, w1b, b1, nullptr, r1b, M, FF, D, 1);
  gemm_bias_kernel<<<dim3(D / 64, M / 128), blk, 0, stream>>>(
      r1b, w2b, b2, ffnf, nullptr, M, D, FF, 0);

  // residual + LN2 -> output
  add_ln_kernel<<<M, 128, 0, stream>>>(h_f, ffnf, g2, be2, (float*)d_out,
                                       nullptr);
}
